// GATModel_30030411333651
// MI455X (gfx1250) — matrix operand order repaired and checked by reading, not compiled, not run on hardware
//
#include <hip/hip_runtime.h>
#include <hip/hip_bf16.h>
#include <math.h>

// Problem constants (match reference)
#define N_NODES 50000
#define N_EDGES 800000
#define FDIM 128           // D_IN = H*D_HID = 128
#define DOUT 16
#define NEG_SLOPE 0.2f

typedef __attribute__((ext_vector_type(16))) __bf16 v16bf;
typedef __attribute__((ext_vector_type(8)))  __bf16 v8bf;
typedef __attribute__((ext_vector_type(8)))  float  v8f;

__device__ __forceinline__ float leaky(float x) { return x >= 0.f ? x : NEG_SLOPE * x; }

// monotonic float <-> uint encoding for atomicMax-based segment max
__device__ __forceinline__ unsigned fkey(float f) {
    unsigned u = __float_as_uint(f);
    return (u & 0x80000000u) ? ~u : (u | 0x80000000u);
}
__device__ __forceinline__ float funkey(unsigned k) {
    return __uint_as_float((k & 0x80000000u) ? (k & 0x7FFFFFFFu) : ~k);
}

// ================= one-time fp32 -> bf16 hi/lo split =================
// hi = bf16(f), lo = bf16(f - hi): 3-term products recover ~fp32 GEMM accuracy.
__global__ void convert_hi_lo_kernel(const float* __restrict__ in, __bf16* __restrict__ hi,
                                     __bf16* __restrict__ lo, long n) {
    long i = (long)blockIdx.x * 256 + threadIdx.x;
    if (i >= n) return;
    float f = in[i];
    __bf16 h = (__bf16)f;           // native cvt (RNE)
    hi[i] = h;
    lo[i] = (__bf16)(f - (float)h);
}

// Split + transpose weights: W[K][n] -> WT_hi/lo[n][K]; makes WMMA B-fragment
// loads fully contiguous (K runs along rows of WT).
__global__ void convert_w_t_kernel(const float* __restrict__ W, __bf16* __restrict__ hiT,
                                   __bf16* __restrict__ loT, int K, int n) {
    int i = blockIdx.x * 256 + threadIdx.x;
    if (i >= K * n) return;
    int k = i / n, c = i % n;
    float f = W[i];
    __bf16 h = (__bf16)f;
    hiT[(size_t)c * K + k] = h;
    loT[(size_t)c * K + k] = (__bf16)(f - (float)h);
}

// ================= WMMA fragment loads (bf16 16x16x32, wave32) =================
// A (16x32): lanes 0-15 row M=lane, K = k0..k0+15;
//            lanes 16-31 row M=lane-16, K = k0+16..k0+31.
// row is 256B-aligned; kb multiples of 8 elems -> 16B-aligned vector loads.
__device__ __forceinline__ v16bf load_a16(const __bf16* __restrict__ row, int kb) {
    v8bf p0 = *(const v8bf*)(row + kb);
    v8bf p1 = *(const v8bf*)(row + kb + 8);
    return __builtin_shufflevector(p0, p1, 0, 1, 2, 3, 4, 5, 6, 7,
                                   8, 9, 10, 11, 12, 13, 14, 15);
}
// B (32x16): lanes 0-15 col N=lane, K = k0..k0+15; lanes 16-31 col N=lane-16,
// K = k0+16..k0+31. With WT[n][K] this is one contiguous 32B load.
__device__ __forceinline__ v16bf load_b16(const __bf16* __restrict__ wrowT, int kb) {
    return *(const v16bf*)(wrowT + kb);
}
// 3-term bf16 split accumulate
__device__ __forceinline__ v8f wmma3(v16bf aH, v16bf aL, v16bf bH, v16bf bL, v8f c) {
    c = __builtin_amdgcn_wmma_f32_16x16x32_bf16(false, aH, false, bH, (short)0, c, false, false);
    c = __builtin_amdgcn_wmma_f32_16x16x32_bf16(false, aL, false, bH, (short)0, c, false, false);
    c = __builtin_amdgcn_wmma_f32_16x16x32_bf16(false, aH, false, bL, (short)0, c, false, false);
    return c;
}

// ---- Projection: fs = X @ Wa, fd = X @ Wb ([N,128] <- [N,128]@[128,128]) ----
// grid = N/16; 8 waves/block, wave w owns output column tile w*16; A reused for both outputs.
__global__ void __launch_bounds__(256)
proj_dual_kernel(const __bf16* __restrict__ Xh, const __bf16* __restrict__ Xl,
                 const __bf16* __restrict__ WaTh, const __bf16* __restrict__ WaTl,
                 const __bf16* __restrict__ WbTh, const __bf16* __restrict__ WbTl,
                 float* __restrict__ outA, float* __restrict__ outB) {
    int lane = threadIdx.x & 31;
    int wave = threadIdx.x >> 5;
    int rowBase = blockIdx.x * 16;
    int col = wave * 16 + (lane & 15);
    const __bf16* xhrow = Xh + (size_t)(rowBase + (lane & 15)) * FDIM;
    const __bf16* xlrow = Xl + (size_t)(rowBase + (lane & 15)) * FDIM;
    const __bf16* waH = WaTh + (size_t)col * FDIM;
    const __bf16* waL = WaTl + (size_t)col * FDIM;
    const __bf16* wbH = WbTh + (size_t)col * FDIM;
    const __bf16* wbL = WbTl + (size_t)col * FDIM;
    v8f cA = {};
    v8f cB = {};
#pragma unroll
    for (int k0 = 0; k0 < FDIM; k0 += 32) {
        int ka = k0 + ((lane & 16) ? 16 : 0);
        int kb = k0 + ((lane & 16) ? 16 : 0);
        v16bf aH = load_a16(xhrow, ka);
        v16bf aL = load_a16(xlrow, ka);
        cA = wmma3(aH, aL, load_b16(waH, kb), load_b16(waL, kb), cA);
        cB = wmma3(aH, aL, load_b16(wbH, kb), load_b16(wbL, kb), cB);
    }
    // D layout: VGPR v -> row rowBase+v (lanes 0-15) / rowBase+v+8 (lanes 16-31)
    int rb = rowBase + ((lane & 16) ? 8 : 0);
#pragma unroll
    for (int v = 0; v < 8; ++v) {
        outA[(size_t)(rb + v) * FDIM + col] = cA[v];
        outB[(size_t)(rb + v) * FDIM + col] = cB[v];
    }
}

// ---- Init aggregation buffer + softmax state ----
__global__ void init_kernel(float* __restrict__ hout, unsigned* __restrict__ mkey,
                            float* __restrict__ ssum) {
    size_t tid = (size_t)blockIdx.x * 256 + threadIdx.x;
    if (tid < (size_t)N_NODES * FDIM) hout[tid] = 0.f;
    if (tid < (size_t)N_NODES * 2) { mkey[tid] = 0u; ssum[tid] = 0.f; }
}

// ---- Edge pass A: logits[e][h] = attn_h . leaky(fs[src]+fd[dst]); atomicMax mkey ----
// one wave per edge; lanes 0-15 = head 0, lanes 16-31 = head 1; float4 row gathers.
__global__ void __launch_bounds__(256)
edge_logits_kernel(const float* __restrict__ fs, const float* __restrict__ fd,
                   const int* __restrict__ src, const int* __restrict__ dst,
                   const float* __restrict__ attn, float* __restrict__ logit,
                   unsigned* __restrict__ mkey) {
    int lane = threadIdx.x & 31;
    int wave = threadIdx.x >> 5;
    int e = blockIdx.x * 8 + wave;
    if (e >= N_EDGES) return;
    int s = src[e], d = dst[e];
    float4 fsv = ((const float4*)(fs + (size_t)s * FDIM))[lane];
    float4 fdv = ((const float4*)(fd + (size_t)d * FDIM))[lane];
    float4 av  = ((const float4*)attn)[lane];   // attn flat [2*64] == [128]
    float p = av.x * leaky(fsv.x + fdv.x)
            + av.y * leaky(fsv.y + fdv.y)
            + av.z * leaky(fsv.z + fdv.z)
            + av.w * leaky(fsv.w + fdv.w);
    p += __shfl_xor(p, 1, 32);
    p += __shfl_xor(p, 2, 32);
    p += __shfl_xor(p, 4, 32);
    p += __shfl_xor(p, 8, 32);                  // reduce within each 16-lane head group
    if ((lane & 15) == 0) {
        int h = lane >> 4;
        logit[(size_t)e * 2 + h] = p;
        atomicMax(&mkey[(size_t)d * 2 + h], fkey(p));
    }
}

// ---- Edge pass B: a = exp(logit - max[dst]); in-place; atomicAdd into ssum ----
__global__ void edge_exp_kernel(float* __restrict__ logit, const int* __restrict__ dst,
                                const unsigned* __restrict__ mkey, float* __restrict__ ssum) {
    size_t i = (size_t)blockIdx.x * 256 + threadIdx.x;
    if (i >= (size_t)N_EDGES * 2) return;
    int e = (int)(i >> 1);
    int h = (int)(i & 1);
    int d = dst[e];
    float m = funkey(mkey[(size_t)d * 2 + h]);
    float a = expf(logit[i] - m);
    logit[i] = a;
    atomicAdd(&ssum[(size_t)d * 2 + h], a);
}

// ---- Edge pass C: hout[dst] += fs[src] * alpha (alpha = a / s[dst]) ----
__global__ void __launch_bounds__(256)
edge_scatter_kernel(const float* __restrict__ fs, const float* __restrict__ aexp,
                    const float* __restrict__ ssum, const int* __restrict__ src,
                    const int* __restrict__ dst, float* __restrict__ hout) {
    int lane = threadIdx.x & 31;
    int wave = threadIdx.x >> 5;
    int e = blockIdx.x * 8 + wave;
    if (e >= N_EDGES) return;
    int s = src[e], d = dst[e];
    int h = lane >> 4;
    float alpha = aexp[(size_t)e * 2 + h] / ssum[(size_t)d * 2 + h];
    float4 fsv = ((const float4*)(fs + (size_t)s * FDIM))[lane];
    float* o = hout + (size_t)d * FDIM + lane * 4;
    atomicAdd(o + 0, fsv.x * alpha);
    atomicAdd(o + 1, fsv.y * alpha);
    atomicAdd(o + 2, fsv.z * alpha);
    atomicAdd(o + 3, fsv.w * alpha);
}

// ---- Finalize: h = leaky(h + bias) ----
__global__ void finalize_kernel(float* __restrict__ h, const float* __restrict__ bias) {
    size_t i = (size_t)blockIdx.x * 256 + threadIdx.x;
    if (i >= (size_t)N_NODES * FDIM) return;
    h[i] = leaky(h[i] + bias[i & (FDIM - 1)]);
}

// ---- Readout: out = h0@R0W + h1@R1W + h2@R2W + (b0+b1+b2), [N,16] ----
__global__ void __launch_bounds__(256)
readout_kernel(const __bf16* __restrict__ h0h, const __bf16* __restrict__ h0l,
               const __bf16* __restrict__ W0h, const __bf16* __restrict__ W0l,
               const __bf16* __restrict__ h1h, const __bf16* __restrict__ h1l,
               const __bf16* __restrict__ W1h, const __bf16* __restrict__ W1l,
               const __bf16* __restrict__ h2h, const __bf16* __restrict__ h2l,
               const __bf16* __restrict__ W2h, const __bf16* __restrict__ W2l,
               const float* __restrict__ b0, const float* __restrict__ b1,
               const float* __restrict__ b2, float* __restrict__ out) {
    int lane = threadIdx.x & 31;
    int wave = threadIdx.x >> 5;
    int t = blockIdx.x * 8 + wave;
    if (t >= N_NODES / 16) return;       // wave-uniform exit; WMMA waves keep full EXEC
    int rowBase = t * 16;
    size_t rowOff = (size_t)(rowBase + (lane & 15)) * FDIM;
    int col = lane & 15;
    size_t wOff = (size_t)col * FDIM;
    const __bf16* Hh[3] = { h0h + rowOff, h1h + rowOff, h2h + rowOff };
    const __bf16* Hl[3] = { h0l + rowOff, h1l + rowOff, h2l + rowOff };
    const __bf16* Wh[3] = { W0h + wOff, W1h + wOff, W2h + wOff };
    const __bf16* Wl[3] = { W0l + wOff, W1l + wOff, W2l + wOff };
    v8f c = {};
#pragma unroll
    for (int m = 0; m < 3; ++m) {
#pragma unroll
        for (int k0 = 0; k0 < FDIM; k0 += 32) {
            int ka = k0 + ((lane & 16) ? 16 : 0);
            int kb = k0 + ((lane & 16) ? 16 : 0);
            v16bf aH = load_a16(Hh[m], ka);
            v16bf aL = load_a16(Hl[m], ka);
            c = wmma3(aH, aL, load_b16(Wh[m], kb), load_b16(Wl[m], kb), c);
        }
    }
    float bsum = b0[col] + b1[col] + b2[col];
    int rb = rowBase + ((lane & 16) ? 8 : 0);
#pragma unroll
    for (int v = 0; v < 8; ++v)
        out[(size_t)(rb + v) * DOUT + col] = c[v] + bsum;
}

extern "C" void kernel_launch(void* const* d_in, const int* in_sizes, int n_in,
                              void* d_out, int out_size, void* d_ws, size_t ws_size,
                              hipStream_t stream) {
    const float* x     = (const float*)d_in[0];
    const int*   src   = (const int*)  d_in[1];
    const int*   dst   = (const int*)  d_in[2];
    const float* Wsrc0 = (const float*)d_in[3];
    const float* Wdst0 = (const float*)d_in[4];
    const float* attn0 = (const float*)d_in[5];
    const float* bias0 = (const float*)d_in[6];
    const float* Wsrc1 = (const float*)d_in[7];
    const float* Wdst1 = (const float*)d_in[8];
    const float* attn1 = (const float*)d_in[9];
    const float* bias1 = (const float*)d_in[10];
    const float* R0W   = (const float*)d_in[11];
    const float* R0b   = (const float*)d_in[12];
    const float* R1W   = (const float*)d_in[13];
    const float* R1b   = (const float*)d_in[14];
    const float* R2W   = (const float*)d_in[15];
    const float* R2b   = (const float*)d_in[16];
    float* out = (float*)d_out;

    const size_t NB = (size_t)N_NODES * FDIM;       // 6.4M elements
    // fp32 region
    float*    fs    = (float*)d_ws;
    float*    fd    = fs + NB;
    float*    h1    = fd + NB;                      // fp32 h1; dead after its bf16 split
    float*    h2    = h1 + NB;
    float*    logit = h2 + NB;                      // E*2 floats, exp'd in place
    unsigned* mkey  = (unsigned*)(logit + (size_t)N_EDGES * 2);
    float*    ssum  = (float*)(mkey + (size_t)N_NODES * 2);
    // bf16 region
    __bf16* xh  = (__bf16*)(ssum + (size_t)N_NODES * 2);
    __bf16* xl  = xh + NB;
    __bf16* h1h = xl + NB;
    __bf16* h1l = h1h + NB;
    __bf16* WaTh = h1l + NB;                        // per-layer weight packs (reused)
    __bf16* WaTl = WaTh + FDIM * FDIM;
    __bf16* WbTh = WaTl + FDIM * FDIM;
    __bf16* WbTl = WbTh + FDIM * FDIM;
    __bf16* R0Th = WbTl + FDIM * FDIM;              // readout weight packs
    __bf16* R0Tl = R0Th + FDIM * DOUT;
    __bf16* R1Th = R0Tl + FDIM * DOUT;
    __bf16* R1Tl = R1Th + FDIM * DOUT;
    __bf16* R2Th = R1Tl + FDIM * DOUT;
    __bf16* R2Tl = R2Th + FDIM * DOUT;
    // alias: h2's bf16 pair overlays the dead h1 fp32 buffer (25.6MB each)
    __bf16* h2h = (__bf16*)h1;
    __bf16* h2l = h2h + NB;

    dim3 blk(256);
    int projGrid = N_NODES / 16;                         // 3125
    int nbGrid   = (int)((NB + 255) / 256);
    int edgeGrid = (N_EDGES + 7) / 8;                    // 1 wave per edge
    int ehGrid   = (int)(((size_t)N_EDGES * 2 + 255) / 256);
    int wGrid    = (FDIM * FDIM + 255) / 256;            // 64
    int rGrid    = (FDIM * DOUT + 255) / 256;            // 8
    int roGrid   = (N_NODES / 16 + 7) / 8;

    // ---- one-time splits for layer 0 ----
    convert_hi_lo_kernel<<<nbGrid, blk, 0, stream>>>(x, xh, xl, (long)NB);
    convert_w_t_kernel<<<wGrid, blk, 0, stream>>>(Wsrc0, WaTh, WaTl, FDIM, FDIM);
    convert_w_t_kernel<<<wGrid, blk, 0, stream>>>(Wdst0, WbTh, WbTl, FDIM, FDIM);
    // ---- layer 0 ----
    proj_dual_kernel<<<projGrid, blk, 0, stream>>>(xh, xl, WaTh, WaTl, WbTh, WbTl, fs, fd);
    init_kernel<<<nbGrid, blk, 0, stream>>>(h1, mkey, ssum);
    edge_logits_kernel<<<edgeGrid, blk, 0, stream>>>(fs, fd, src, dst, attn0, logit, mkey);
    edge_exp_kernel<<<ehGrid, blk, 0, stream>>>(logit, dst, mkey, ssum);
    edge_scatter_kernel<<<edgeGrid, blk, 0, stream>>>(fs, logit, ssum, src, dst, h1);
    finalize_kernel<<<nbGrid, blk, 0, stream>>>(h1, bias0);
    // ---- splits for layer 1 ----
    convert_hi_lo_kernel<<<nbGrid, blk, 0, stream>>>(h1, h1h, h1l, (long)NB);
    convert_w_t_kernel<<<wGrid, blk, 0, stream>>>(Wsrc1, WaTh, WaTl, FDIM, FDIM);
    convert_w_t_kernel<<<wGrid, blk, 0, stream>>>(Wdst1, WbTh, WbTl, FDIM, FDIM);
    // ---- layer 1 ----
    proj_dual_kernel<<<projGrid, blk, 0, stream>>>(h1h, h1l, WaTh, WaTl, WbTh, WbTl, fs, fd);
    init_kernel<<<nbGrid, blk, 0, stream>>>(h2, mkey, ssum);
    edge_logits_kernel<<<edgeGrid, blk, 0, stream>>>(fs, fd, src, dst, attn1, logit, mkey);
    edge_exp_kernel<<<ehGrid, blk, 0, stream>>>(logit, dst, mkey, ssum);
    edge_scatter_kernel<<<edgeGrid, blk, 0, stream>>>(fs, logit, ssum, src, dst, h2);
    finalize_kernel<<<nbGrid, blk, 0, stream>>>(h2, bias1);
    // ---- readout splits (h2 bf16 pair overlays dead h1 fp32) ----
    convert_hi_lo_kernel<<<nbGrid, blk, 0, stream>>>(h2, h2h, h2l, (long)NB);
    convert_w_t_kernel<<<rGrid, blk, 0, stream>>>(R0W, R0Th, R0Tl, FDIM, DOUT);
    convert_w_t_kernel<<<rGrid, blk, 0, stream>>>(R1W, R1Th, R1Tl, FDIM, DOUT);
    convert_w_t_kernel<<<rGrid, blk, 0, stream>>>(R2W, R2Th, R2Tl, FDIM, DOUT);
    // ---- jumping-knowledge readout ----
    readout_kernel<<<roGrid, blk, 0, stream>>>(xh, xl, R0Th, R0Tl,
                                               h1h, h1l, R1Th, R1Tl,
                                               h2h, h2l, R2Th, R2Tl,
                                               R0b, R1b, R2b, out);

    (void)in_sizes; (void)n_in; (void)out_size; (void)ws_size;
}
